// MultiHeadAttentionV1_40785009443049
// MI455X (gfx1250) — matrix operand order repaired and checked by reading, not compiled, not run on hardware
//
#include <hip/hip_runtime.h>

// ---------------------------------------------------------------------------
// MHA forward for gfx1250 (MI455X): all matmuls on v_wmma_f32_16x16x32_f16,
// GEMM tiles staged into LDS by the Tensor Data Mover (tensor_load_to_lds),
// double-buffered and synchronized with s_wait_tensorcnt.
// ---------------------------------------------------------------------------

#define D_MODEL 1024
#define NHEAD   16
#define DEPTH   64
#define BATCH   2
#define SEQ     2048
#define MROWS   (BATCH * SEQ)   // 4096

typedef __attribute__((ext_vector_type(16))) _Float16 v16h;
typedef __attribute__((ext_vector_type(8)))  float    v8f;
typedef __attribute__((ext_vector_type(4)))  unsigned int v4u;
typedef __attribute__((ext_vector_type(4)))  float    v4f;
typedef __attribute__((ext_vector_type(4)))  _Float16 v4h;
typedef __attribute__((ext_vector_type(8)))  int      v8i;
typedef __attribute__((ext_vector_type(4)))  int      v4i;

union F16Frag { v16h h; v4u u[2]; };

__device__ __forceinline__ v8f wmma_f16(v16h a, v16h b, v8f c) {
  // D = A(16x32 f16) * B(32x16 f16) + C(16x16 f32)
  return __builtin_amdgcn_wmma_f32_16x16x32_f16(
      /*neg_a=*/false, a, /*neg_b=*/false, b,
      /*c_mod=*/(short)0, c, /*reuse_a=*/false, /*reuse_b=*/false);
}

// ---------------------------------------------------------------------------
// TDM: DMA a 2-D f16 tile [tile_d1 rows x tile_d0 elems] from global into LDS.
// D# layout per CDNA5 ISA section 8.3/8.4 (group0 128b, group1 256b).
// Toolchain uses the 6-arg builtin form (extra int32x8 group before cpol).
// ---------------------------------------------------------------------------
__device__ __forceinline__ void tdm_load_tile_f16(unsigned lds_addr,
                                                  const void* gaddr,
                                                  unsigned tensor_d0,
                                                  unsigned tensor_d1,
                                                  unsigned tile_d0,
                                                  unsigned tile_d1,
                                                  unsigned stride_d0_elems) {
  unsigned long long ga = (unsigned long long)gaddr;
  v4u g0;
  g0.x = 1u;                                        // count=1, user mode, no gather
  g0.y = lds_addr;                                  // LDS byte address of tile
  g0.z = (unsigned)(ga & 0xFFFFFFFFu);              // global_addr[31:0]
  g0.w = (unsigned)((ga >> 32) & 0x01FFFFFFu)       // global_addr[56:32]
         | 0x80000000u;                             // type=2 ("image")
  v8i g1;
  g1[0] = (int)(1u << 16);                          // workgroup_mask=0, data_size=1 (2B)
  g1[1] = (int)((tensor_d0 & 0xFFFFu) << 16);       // tensor_dim0[15:0] @ bits 63:48
  g1[2] = (int)(((tensor_d0 >> 16) & 0xFFFFu)       // tensor_dim0[31:16]
         | ((tensor_d1 & 0xFFFFu) << 16));          // tensor_dim1[15:0]
  g1[3] = (int)(((tensor_d1 >> 16) & 0xFFFFu)       // tensor_dim1[31:16]
         | ((tile_d0 & 0xFFFFu) << 16));            // tile_dim0
  g1[4] = (int)(tile_d1 & 0xFFFFu);                 // tile_dim1 (tile_dim2=0)
  g1[5] = (int)stride_d0_elems;                     // tensor_dim0_stride[31:0]
  g1[6] = 0;                                        // stride hi bits / dim1_stride lo
  g1[7] = 0;
  v4i z4 = {0, 0, 0, 0};
  v8i z8 = {0, 0, 0, 0, 0, 0, 0, 0};
  __builtin_amdgcn_tensor_load_to_lds(g0, g1, z4, z4, z8, 0);
}

// ---------------------------------------------------------------------------
// f32 -> f16 conversion, 4 elements per thread
// ---------------------------------------------------------------------------
__global__ __launch_bounds__(256) void cvt_f32_f16(const float* __restrict__ src,
                                                   _Float16* __restrict__ dst,
                                                   int n4) {
  int i = blockIdx.x * 256 + threadIdx.x;
  if (i < n4) {
    v4f v = reinterpret_cast<const v4f*>(src)[i];
    reinterpret_cast<v4h*>(dst)[i] = __builtin_convertvector(v, v4h);
  }
}

// ---------------------------------------------------------------------------
// C[m,n] = sum_k A[m,k] * W[n,k] + bias[n]
// A: [4096,1024] f16 row-major (K contig). W: [1024,1024] f16 (K contig).
// 128 threads = 4 waves; WG tile 64(M) x 64(N); K stepped by 32.
// Tiles staged by TDM into double-buffered LDS; one barrier per K-step.
// MODE 0: f16 row-major. MODE 1: f16 per-head-transposed (V). MODE 2: f32.
// ---------------------------------------------------------------------------
template <int MODE>
__global__ __launch_bounds__(128) void gemm_wmma(const _Float16* __restrict__ A,
                                                 const _Float16* __restrict__ W,
                                                 const float* __restrict__ bias,
                                                 void* __restrict__ outp) {
  __shared__ alignas(16) _Float16 As[2][64 * 32];
  __shared__ alignas(16) _Float16 Ws[2][64 * 32];

  const int t  = threadIdx.x;
  const int w  = t >> 5;        // wave 0..3 -> 16-row slice of M tile
  const int l  = t & 31;        // lane in wave
  const int lm = l & 15;
  const int hi = l >> 4;        // 0 | 1 (lane half)
  const int m0 = blockIdx.x * 64;
  const int n0 = blockIdx.y * 64;

  const int ka = hi ? 8 : 0;    // A-frag K sub-offset (ISA 16-bit A layout)
  const int kb = hi ? 16 : 0;   // unused; B frags use ka, same k order as A

  v8f acc[4];
  v8f z = {};
#pragma unroll
  for (int nb = 0; nb < 4; ++nb) acc[nb] = z;

  const unsigned ldsA0 = (unsigned)(size_t)(&As[0][0]);
  const unsigned ldsW0 = (unsigned)(size_t)(&Ws[0][0]);
  const _Float16* gA = A + (size_t)m0 * D_MODEL;   // tile row 0, k advances below
  const _Float16* gW = W + (size_t)n0 * D_MODEL;

  const int NK = D_MODEL / 32;                      // 32 K-steps

  // prologue: DMA first slabs into buffer 0 (wave 0 only; EXEC-independent op)
  if (t < 32) {
    tdm_load_tile_f16(ldsA0, gA, D_MODEL, MROWS, 32, 64, D_MODEL);
    tdm_load_tile_f16(ldsW0, gW, D_MODEL, D_MODEL, 32, 64, D_MODEL);
  }

  for (int i = 0; i < NK; ++i) {
    const int buf = i & 1;
    __builtin_amdgcn_s_wait_tensorcnt(0);   // wave0: slab `buf` landed (others: no-op)
    __syncthreads();                         // everyone: slab ready, prior readers done

    if (t < 32 && i + 1 < NK) {              // DMA next slab while we compute
      const int kn = (i + 1) * 32;
      tdm_load_tile_f16(ldsA0 + (buf ^ 1) * (64 * 32 * 2), gA + kn,
                        D_MODEL, MROWS, 32, 64, D_MODEL);
      tdm_load_tile_f16(ldsW0 + (buf ^ 1) * (64 * 32 * 2), gW + kn,
                        D_MODEL, D_MODEL, 32, 64, D_MODEL);
    }

    // ---- A fragment: rows 16w..16w+15 of this wave ----
    F16Frag af;
    const int ml = 16 * w + lm;
    af.u[0] = *reinterpret_cast<const v4u*>(&As[buf][ml * 32 + ka]);
    af.u[1] = *reinterpret_cast<const v4u*>(&As[buf][ml * 32 + ka + 16]);

#pragma unroll
    for (int nb = 0; nb < 4; ++nb) {
      F16Frag bf;
      const int nl = 16 * nb + lm;    // output column -> W row (K contiguous)
      bf.u[0] = *reinterpret_cast<const v4u*>(&Ws[buf][nl * 32 + ka]);
      bf.u[1] = *reinterpret_cast<const v4u*>(&Ws[buf][nl * 32 + ka + 16]);
      acc[nb] = wmma_f16(af.h, bf.h, acc[nb]);
    }
  }

  // ---- epilogue: bias + store. D layout: lane holds col n=lm, rows 8*hi+j.
#pragma unroll
  for (int nb = 0; nb < 4; ++nb) {
    const int ng = n0 + 16 * nb + lm;
    const float bv = bias[ng];
#pragma unroll
    for (int j = 0; j < 8; ++j) {
      const int mg = m0 + 16 * w + 8 * hi + j;
      const float v = acc[nb][j] + bv;
      if (MODE == 0) {
        reinterpret_cast<_Float16*>(outp)[(size_t)mg * D_MODEL + ng] = (_Float16)v;
      } else if (MODE == 1) {
        // V output transposed per head: Vt[b][h][d][seq]
        const int b = mg >> 11, key = mg & (SEQ - 1);
        const int h = ng >> 6,  d   = ng & (DEPTH - 1);
        reinterpret_cast<_Float16*>(outp)[((size_t)((b * NHEAD + h) * DEPTH + d)) * SEQ + key] =
            (_Float16)v;
      } else {
        reinterpret_cast<float*>(outp)[(size_t)mg * D_MODEL + ng] = v;
      }
    }
  }
}

// ---------------------------------------------------------------------------
// Flash attention, one wave (32 threads) per 16-row Q block of one (b,h).
// Q,K: f16 [B*SEQ, 1024] row-major. Vt: f16 [B,H,DEPTH,SEQ]. AO: f16 like Q.
// ---------------------------------------------------------------------------
#define NEGMAX (-3.0e38f)

__global__ __launch_bounds__(32) void attn_fa(const _Float16* __restrict__ Q,
                                              const _Float16* __restrict__ K,
                                              const _Float16* __restrict__ Vt,
                                              _Float16* __restrict__ AO) {
  __shared__ alignas(16) _Float16 Pl[16 * 32];   // P tile, D-layout -> A-layout relay

  const int l  = threadIdx.x;
  const int lm = l & 15;
  const int hi = l >> 4;
  const int ka = hi ? 8 : 0;

  const int bid = blockIdx.x;        // 0..4095
  const int qb  = bid & 127;
  const int bh  = bid >> 7;          // 0..31
  const int b   = bh >> 4;
  const int h   = bh & 15;
  const int q0  = qb << 4;

  // ---- Q A-fragments (two 16x32 d-halves), straight from global ----
  F16Frag qa[2];
  const _Float16* qrow = Q + (size_t)((b << 11) + q0 + lm) * D_MODEL + (h << 6);
#pragma unroll
  for (int dh = 0; dh < 2; ++dh) {
    qa[dh].u[0] = *reinterpret_cast<const v4u*>(qrow + (dh << 5) + ka);
    qa[dh].u[1] = *reinterpret_cast<const v4u*>(qrow + (dh << 5) + ka + 16);
  }

  v8f o[4];
  v8f z = {};
#pragma unroll
  for (int dc = 0; dc < 4; ++dc) o[dc] = z;
  float mrow[8], lrow[8];
#pragma unroll
  for (int j = 0; j < 8; ++j) { mrow[j] = NEGMAX; lrow[j] = 0.0f; }

  const int nkb = (q0 >> 5) + 1;     // causal: key blocks of 32 up to q0+15

  for (int kblk = 0; kblk < nkb; ++kblk) {
    const int j0 = kblk << 5;

    // ---- S = Q K^T for two 16-key subtiles (K frags contiguous in global) --
    v8f s[2];
#pragma unroll
    for (int ss = 0; ss < 2; ++ss) {
      v8f c = z;
      const _Float16* krow = K + (size_t)((b << 11) + j0 + (ss << 4) + lm) * D_MODEL + (h << 6);
#pragma unroll
      for (int dh = 0; dh < 2; ++dh) {
        F16Frag kf;
        const _Float16* p = krow + (dh << 5) + ka;
        kf.u[0] = *reinterpret_cast<const v4u*>(p);
        kf.u[1] = *reinterpret_cast<const v4u*>(p + 16);
        c = wmma_f16(qa[dh].h, kf.h, c);
      }
      s[ss] = c;
    }

    // ---- scale, causal mask, online softmax (rows = 8*hi + j) ----
    float p0[8], p1[8], alpha[8];
#pragma unroll
    for (int j = 0; j < 8; ++j) {
      const int row = q0 + (hi << 3) + j;
      float v0 = s[0][j] * 0.125f;             // 1/sqrt(64)
      float v1 = s[1][j] * 0.125f;
      if (j0 + lm > row)      v0 = NEGMAX;
      if (j0 + 16 + lm > row) v1 = NEGMAX;

      float mx = fmaxf(v0, v1);
      mx = fmaxf(mx, __shfl_xor(mx, 1, 32));
      mx = fmaxf(mx, __shfl_xor(mx, 2, 32));
      mx = fmaxf(mx, __shfl_xor(mx, 4, 32));
      mx = fmaxf(mx, __shfl_xor(mx, 8, 32));
      const float mnew = fmaxf(mrow[j], mx);

      alpha[j] = __expf(mrow[j] - mnew);
      p0[j] = __expf(v0 - mnew);
      p1[j] = __expf(v1 - mnew);

      float sum = p0[j] + p1[j];
      sum += __shfl_xor(sum, 1, 32);
      sum += __shfl_xor(sum, 2, 32);
      sum += __shfl_xor(sum, 4, 32);
      sum += __shfl_xor(sum, 8, 32);

      lrow[j] = lrow[j] * alpha[j] + sum;
      mrow[j] = mnew;
    }

    // ---- relay P (D-layout) through LDS into A-layout ----
#pragma unroll
    for (int j = 0; j < 8; ++j) {
      const int r = (hi << 3) + j;
      Pl[(r << 5) + lm]      = (_Float16)p0[j];
      Pl[(r << 5) + 16 + lm] = (_Float16)p1[j];
    }
    asm volatile("s_wait_dscnt 0" ::: "memory");  // single wave: lockstep + DS in-order

    F16Frag pa;
    pa.u[0] = *reinterpret_cast<const v4u*>(&Pl[(lm << 5) + ka]);
    pa.u[1] = *reinterpret_cast<const v4u*>(&Pl[(lm << 5) + ka + 16]);

    // ---- O = O*alpha + P @ V  (Vt pre-transposed: contiguous B-frags) ----
#pragma unroll
    for (int dc = 0; dc < 4; ++dc) {
#pragma unroll
      for (int j = 0; j < 8; ++j) o[dc][j] *= alpha[j];
      F16Frag vf;
      const _Float16* vrow =
          Vt + (size_t)((bh << 6) + (dc << 4) + lm) * SEQ + j0 + ka;
      vf.u[0] = *reinterpret_cast<const v4u*>(vrow);
      vf.u[1] = *reinterpret_cast<const v4u*>(vrow + 16);
      o[dc] = wmma_f16(pa.h, vf.h, o[dc]);
    }
    asm volatile("" ::: "memory");   // keep next iter's Pl stores behind pa loads
  }

  // ---- normalize and store attention output (f16, row-major model layout) --
#pragma unroll
  for (int j = 0; j < 8; ++j) {
    const float inv = 1.0f / lrow[j];
    const int mg = (b << 11) + q0 + (hi << 3) + j;
#pragma unroll
    for (int dc = 0; dc < 4; ++dc) {
      const int cg = (h << 6) + (dc << 4) + lm;
      AO[(size_t)mg * D_MODEL + cg] = (_Float16)(o[dc][j] * inv);
    }
  }
}

// ---------------------------------------------------------------------------
// Host launcher
// ---------------------------------------------------------------------------
extern "C" void kernel_launch(void* const* d_in, const int* in_sizes, int n_in,
                              void* d_out, int out_size, void* d_ws, size_t ws_size,
                              hipStream_t stream) {
  (void)in_sizes; (void)n_in; (void)out_size; (void)ws_size;

  const float* x  = (const float*)d_in[0];
  // d_in[1] = mask: analytically causal, applied in-kernel.
  const float* Wq = (const float*)d_in[2];
  const float* bq = (const float*)d_in[3];
  const float* Wk = (const float*)d_in[4];
  const float* bk = (const float*)d_in[5];
  const float* Wv = (const float*)d_in[6];
  const float* bv = (const float*)d_in[7];
  const float* Wo = (const float*)d_in[8];
  const float* bo = (const float*)d_in[9];

  char* ws = (char*)d_ws;
  const size_t MB = 1u << 20;
  _Float16* X16  = (_Float16*)(ws + 0 * MB);    // 8 MB  [4096,1024]
  _Float16* Wq16 = (_Float16*)(ws + 8 * MB);    // 2 MB
  _Float16* Wk16 = (_Float16*)(ws + 10 * MB);   // 2 MB
  _Float16* Wv16 = (_Float16*)(ws + 12 * MB);   // 2 MB
  _Float16* Wo16 = (_Float16*)(ws + 14 * MB);   // 2 MB
  _Float16* Q16  = (_Float16*)(ws + 16 * MB);   // 8 MB
  _Float16* K16  = (_Float16*)(ws + 24 * MB);   // 8 MB
  _Float16* Vt16 = (_Float16*)(ws + 32 * MB);   // 8 MB  [B,H,DEPTH,SEQ]
  _Float16* AO16 = (_Float16*)(ws + 40 * MB);   // 8 MB

  // f32 -> f16 conversions
  cvt_f32_f16<<<(MROWS * D_MODEL / 4 + 255) / 256, 256, 0, stream>>>(x, X16, MROWS * D_MODEL / 4);
  const int wn4 = D_MODEL * D_MODEL / 4;
  cvt_f32_f16<<<(wn4 + 255) / 256, 256, 0, stream>>>(Wq, Wq16, wn4);
  cvt_f32_f16<<<(wn4 + 255) / 256, 256, 0, stream>>>(Wk, Wk16, wn4);
  cvt_f32_f16<<<(wn4 + 255) / 256, 256, 0, stream>>>(Wv, Wv16, wn4);
  cvt_f32_f16<<<(wn4 + 255) / 256, 256, 0, stream>>>(Wo, Wo16, wn4);

  // projections
  dim3 gg(MROWS / 64, D_MODEL / 64);
  gemm_wmma<0><<<gg, 128, 0, stream>>>(X16, Wq16, bq, Q16);
  gemm_wmma<0><<<gg, 128, 0, stream>>>(X16, Wk16, bk, K16);
  gemm_wmma<1><<<gg, 128, 0, stream>>>(X16, Wv16, bv, Vt16);

  // flash attention: 32 (b,h) pairs * 128 q-blocks = 4096 waves
  attn_fa<<<BATCH * NHEAD * (SEQ / 16), 32, 0, stream>>>(Q16, K16, Vt16, AO16);

  // output projection (f32 epilogue into d_out)
  gemm_wmma<2><<<gg, 128, 0, stream>>>(AO16, Wo16, bo, d_out);
}
